// BidirectionalRNN_ANN_28905129902329
// MI455X (gfx1250) — matrix operand order repaired and checked by reading, not compiled, not run on hardware
//
#include <hip/hip_runtime.h>
#include <hip/hip_bf16.h>

// ---------------- problem constants ----------------
#define SEQ  512
#define BB   256          // batch
#define INF_ 300          // input features
#define KP   320          // padded K for x@Wx (multiple of 32)
#define HH   256          // hidden
#define H2   512          // 2*H
#define MSEQ (SEQ * BB)   // rows of the big projection GEMM

typedef __attribute__((ext_vector_type(16))) _Float16 v16h;
typedef __attribute__((ext_vector_type(8)))  float    v8f;
typedef __attribute__((ext_vector_type(4)))  int      v4i;

// LDS address-space typedefs for async LDS copies
typedef __attribute__((address_space(3))) void lvoid;   // LDS void
typedef __attribute__((address_space(3))) v4i  lv4i;    // LDS int4

#define HAVE_ASYNC_LDS __has_builtin(__builtin_amdgcn_global_load_async_to_lds_b128)

__device__ __forceinline__ void wait_async0() {
#if __has_builtin(__builtin_amdgcn_s_wait_asynccnt)
    __builtin_amdgcn_s_wait_asynccnt(0);
#else
    asm volatile("s_wait_asynccnt 0x0" ::: "memory");
#endif
}

// fast tanh via v_exp_f32 (transcendental VALU, off the WMMA pipe)
__device__ __forceinline__ float ftanh(float x) {
    float e = __expf(2.0f * x);
    return 1.0f - 2.0f / (e + 1.0f);
}

// ---- WMMA fragment loaders (contiguous k order, same for A and B) ----
// A (16x32 f16): lane m=l&15 is row M; v[0..15] = K(base..base+15)
// with base = (l>>4)*16.
__device__ __forceinline__ v16h load_a_frag(const _Float16* __restrict__ tile,
                                            int ld, int lane) {
    int m  = lane & 15;
    int ko = (lane >> 4) * 16;
    const _Float16* p = tile + (size_t)m * ld + ko;
    v16h r;
#pragma unroll
    for (int j = 0; j < 8; ++j) { r[j] = p[j]; r[j + 8] = p[j + 8]; }
    return r;
}

// B (32x16 f16) from K-transposed weights Wt[n][k] (contiguous in k):
// lane n=l&15 is column N; v[0..15] = K(kb..kb+15), kb = (l>>4)*16.
__device__ __forceinline__ v16h load_b_frag_t(const _Float16* __restrict__ wt,
                                              int ld, int lane) {
    int n  = lane & 15;
    int kb = (lane >> 4) * 16;
    const _Float16* p = wt + (size_t)n * ld + kb;
    v16h r;
#pragma unroll
    for (int j = 0; j < 16; ++j) r[j] = p[j];
    return r;
}

// ---------------- conversion kernels ----------------
// seq f32 [SEQ,B,IN] -> Xh f16 [SEQ*B, KP] zero-padded in K
__global__ void cvt_x_kernel(const float* __restrict__ seq,
                             _Float16* __restrict__ Xh) {
    size_t i = (size_t)blockIdx.x * blockDim.x + threadIdx.x;
    if (i >= (size_t)MSEQ * KP) return;
    size_t row = i / KP;
    int    k   = (int)(i % KP);
    Xh[i] = (k < INF_) ? (_Float16)seq[row * INF_ + k] : (_Float16)0.0f;
}

// Wx f32 [IN,H] -> Wt f16 [H, KP] (transposed, zero-padded) for both dirs
__global__ void cvt_w_kernel(const float* __restrict__ W1x,
                             const float* __restrict__ W2x,
                             _Float16* __restrict__ W1t,
                             _Float16* __restrict__ W2t) {
    int i = blockIdx.x * blockDim.x + threadIdx.x;   // 2*H*KP threads
    if (i >= 2 * HH * KP) return;
    int dir = i / (HH * KP);
    int r   = i % (HH * KP);
    int n   = r / KP;
    int k   = r % KP;
    const float* W = dir ? W2x : W1x;
    _Float16*    T = dir ? W2t : W1t;
    T[r] = (k < INF_) ? (_Float16)W[(size_t)k * HH + n] : (_Float16)0.0f;
}

// ---------------- big projection GEMM: px = Xh @ Wt^T + bias (f16 out) ----
// one wave per 16x16 output tile; K = KP = 320 (10 chunks of 32)
__global__ void gemm_px_kernel(const _Float16* __restrict__ X,
                               const _Float16* __restrict__ Wt,
                               const float* __restrict__ bias,
                               _Float16* __restrict__ out) {
    int wid  = (int)((blockIdx.x * blockDim.x + threadIdx.x) >> 5);
    int lane = threadIdx.x & 31;
    int nt = wid & 15;          // H/16 = 16 column tiles
    int mt = wid >> 4;          // MSEQ/16 = 8192 row tiles
    if (mt >= MSEQ / 16) return;

    const _Float16* Arow = X  + (size_t)mt * 16 * KP;
    const _Float16* Wrow = Wt + (size_t)nt * 16 * KP;

    v8f acc = {};
#pragma unroll
    for (int kc = 0; kc < KP / 32; ++kc) {
        v16h a = load_a_frag(Arow + kc * 32, KP, lane);
        v16h b = load_b_frag_t(Wrow + kc * 32, KP, lane);
        acc = __builtin_amdgcn_wmma_f32_16x16x32_f16(
                  false, a, false, b, (short)0, acc, false, false);
    }
    int mrow = (lane >> 4) * 8;
    int n    = nt * 16 + (lane & 15);
    float bv = bias[n];
#pragma unroll
    for (int j = 0; j < 8; ++j)
        out[(size_t)(mt * 16 + mrow + j) * HH + n] = (_Float16)(acc[j] + bv);
}

// ---------------- sequential recurrence ----------------
// grid (B/16, 2 dirs); block = 256 threads = 8 waves.
// Block owns a 16-row batch slice; wave w owns column tiles 2w, 2w+1.
// Wh B-fragments (2 tiles x 8 kchunks x 8 VGPRs = 128 VGPRs) stay resident
// in registers for all 512 steps; h slice (16x256 f16) lives in LDS.
// px slabs (8KB/step) are double-buffered into LDS with async copies.
__global__ void __launch_bounds__(256, 1)
rnn_scan_kernel(const _Float16* __restrict__ px1,
                const _Float16* __restrict__ px2,
                const float* __restrict__ W1h,
                const float* __restrict__ W2h,
                _Float16* __restrict__ hcat) {
    const int dir = blockIdx.y;
    const _Float16* px = dir ? px2 : px1;
    const float*    Wh = dir ? W2h : W1h;
    const int b0   = blockIdx.x * 16;
    const int wave = threadIdx.x >> 5;
    const int lane = threadIdx.x & 31;
    const int c0   = wave * 2;          // first of two column tiles

    __shared__ _Float16 hbuf[16][HH + 8];   // +8 halves pad -> 4-bank skew
#if HAVE_ASYNC_LDS
    __shared__ _Float16 pxbuf[2][16 * HH];  // double-buffered px slab (2x8KB)
#endif

    // --- resident Wh B-fragments ---
    const int n  = lane & 15;
    const int kb = (lane >> 4) * 16;
    v16h bfrag[2][8];
#pragma unroll
    for (int t = 0; t < 2; ++t) {
        int col = (c0 + t) * 16 + n;
#pragma unroll
        for (int kc = 0; kc < 8; ++kc) {
#pragma unroll
            for (int j = 0; j < 16; ++j)
                bfrag[t][kc][j] = (_Float16)Wh[(size_t)(kc * 32 + kb + j) * HH + col];
        }
    }

    // h0 = 0
    for (int i = threadIdx.x; i < 16 * (HH + 8); i += 256)
        ((_Float16*)hbuf)[i] = (_Float16)0.0f;

#if HAVE_ASYNC_LDS
    // prologue: async-copy the first px slab into buffer 0
    {
        const int t0 = dir ? (SEQ - 1) : 0;
        const _Float16* g = px + ((size_t)t0 * BB + b0) * HH;
#pragma unroll
        for (int k = 0; k < 2; ++k) {
            int i = (int)threadIdx.x + k * 256;     // 512 chunks of 16B
            __builtin_amdgcn_global_load_async_to_lds_b128(
                (v4i*)(void*)(g + i * 8),
                (lv4i*)(lvoid*)(&pxbuf[0][i * 8]), 0, 0);
        }
        wait_async0();
    }
#endif
    __syncthreads();

    const int m    = lane & 15;
    const int ko   = (lane >> 4) * 16;
    const int mrow = (lane >> 4) * 8;
    int cur = 0;
    (void)cur;

    for (int s = 0; s < SEQ; ++s) {
        const int t = dir ? (SEQ - 1 - s) : s;

#if HAVE_ASYNC_LDS
        // kick off async copy of next step's slab into the other buffer
        if (s + 1 < SEQ) {
            const int tn = dir ? (t - 1) : (t + 1);
            const _Float16* g = px + ((size_t)tn * BB + b0) * HH;
            _Float16* dst = &pxbuf[cur ^ 1][0];
#pragma unroll
            for (int k = 0; k < 2; ++k) {
                int i = (int)threadIdx.x + k * 256;
                __builtin_amdgcn_global_load_async_to_lds_b128(
                    (v4i*)(void*)(g + i * 8),
                    (lv4i*)(lvoid*)(dst + i * 8), 0, 0);
            }
        }
        const _Float16* pxt = &pxbuf[cur][0];       // LDS-resident slab
#else
        const _Float16* pxt = px + ((size_t)t * BB + b0) * HH;
        if (s + 1 < SEQ) {
            const int tn = dir ? (t - 1) : (t + 1);
            const char* pn = (const char*)(px + ((size_t)tn * BB + b0) * HH);
            __builtin_prefetch(pn + (size_t)threadIdx.x * 32, 0, 1);
        }
#endif

        // C init = px[t] tile (bias already folded in)
        v8f acc0, acc1;
#pragma unroll
        for (int j = 0; j < 8; ++j) {
            acc0[j] = (float)pxt[(size_t)(mrow + j) * HH + c0 * 16 + n];
            acc1[j] = (float)pxt[(size_t)(mrow + j) * HH + (c0 + 1) * 16 + n];
        }

        // h @ Wh : 8 k-chunks, A shared between the wave's two tiles
#pragma unroll
        for (int kc = 0; kc < 8; ++kc) {
            const _Float16* p = &hbuf[m][kc * 32 + ko];
            v16h a;
#pragma unroll
            for (int j = 0; j < 8; ++j) { a[j] = p[j]; a[j + 8] = p[j + 8]; }
            acc0 = __builtin_amdgcn_wmma_f32_16x16x32_f16(
                       false, a, false, bfrag[0][kc], (short)0, acc0, false, false);
            acc1 = __builtin_amdgcn_wmma_f32_16x16x32_f16(
                       false, a, false, bfrag[1][kc], (short)0, acc1, false, false);
        }

#pragma unroll
        for (int j = 0; j < 8; ++j) { acc0[j] = ftanh(acc0[j]); acc1[j] = ftanh(acc1[j]); }

        __syncthreads();                       // all reads of h_{t-1} done
#pragma unroll
        for (int j = 0; j < 8; ++j) {
            hbuf[mrow + j][c0 * 16 + n]       = (_Float16)acc0[j];
            hbuf[mrow + j][(c0 + 1) * 16 + n] = (_Float16)acc1[j];
        }

#if HAVE_ASYNC_LDS
        wait_async0();                         // this wave's async copy done
#endif
        __syncthreads();                       // h_t + next slab visible

#if HAVE_ASYNC_LDS
        cur ^= 1;
#endif

        if (s == SEQ - 1) {                    // emit final hidden into concat
            const int colOff = dir ? HH : 0;
#pragma unroll
            for (int j = 0; j < 8; ++j) {
                hcat[(size_t)(b0 + mrow + j) * H2 + colOff + c0 * 16 + n]
                    = (_Float16)acc0[j];
                hcat[(size_t)(b0 + mrow + j) * H2 + colOff + (c0 + 1) * 16 + n]
                    = (_Float16)acc1[j];
            }
        }
    }
}

// ---------------- MLP head GEMM: out = act(A @ W + b), W f32 [K,N] --------
__global__ void gemm_fc_kernel(const _Float16* __restrict__ A,
                               const float* __restrict__ W,
                               const float* __restrict__ bias,
                               _Float16* __restrict__ out,
                               int Ntiles, int K, int N, int totalTiles,
                               int relu) {
    int wid  = (int)((blockIdx.x * blockDim.x + threadIdx.x) >> 5);
    int lane = threadIdx.x & 31;
    if (wid >= totalTiles) return;
    int nt = wid % Ntiles;
    int mt = wid / Ntiles;

    const int n  = lane & 15;
    const int kb = (lane >> 4) * 16;

    v8f acc = {};
    for (int kc = 0; kc < K / 32; ++kc) {
        v16h a = load_a_frag(A + (size_t)mt * 16 * K + kc * 32, K, lane);
        // B fragment with on-the-fly f32 -> f16 conversion (column n strided)
        v16h b;
        const float* wp = W + (size_t)(kc * 32 + kb) * N + nt * 16 + n;
#pragma unroll
        for (int j = 0; j < 16; ++j) b[j] = (_Float16)wp[(size_t)j * N];
        acc = __builtin_amdgcn_wmma_f32_16x16x32_f16(
                  false, a, false, b, (short)0, acc, false, false);
    }
    int mrow = (lane >> 4) * 8;
    int col  = nt * 16 + n;
    float bv = bias[col];
#pragma unroll
    for (int j = 0; j < 8; ++j) {
        float v = acc[j] + bv;
        if (relu && v < 0.0f) v = 0.0f;
        out[(size_t)(mt * 16 + mrow + j) * N + col] = (_Float16)v;
    }
}

// ---------------- final scalar head: out[b] = tanh(y2[b,:] . fs_w + fs_b) --
__global__ void fs_head_kernel(const _Float16* __restrict__ y2,
                               const float* __restrict__ fsw,
                               const float* __restrict__ fsb,
                               float* __restrict__ out) {
    int b = blockIdx.x * blockDim.x + threadIdx.x;
    if (b >= BB) return;
    float s = fsb[0];
#pragma unroll 8
    for (int k = 0; k < HH; ++k)
        s += (float)y2[(size_t)b * HH + k] * fsw[k];
    out[b] = ftanh(s);
}

// ---------------- host orchestration ----------------
extern "C" void kernel_launch(void* const* d_in, const int* in_sizes, int n_in,
                              void* d_out, int out_size, void* d_ws, size_t ws_size,
                              hipStream_t stream) {
    (void)in_sizes; (void)n_in; (void)out_size; (void)ws_size;
    const float* seq   = (const float*)d_in[0];
    const float* W1x   = (const float*)d_in[1];
    const float* W1h   = (const float*)d_in[2];
    const float* b1    = (const float*)d_in[3];
    const float* W2x   = (const float*)d_in[4];
    const float* W2h   = (const float*)d_in[5];
    const float* b2    = (const float*)d_in[6];
    const float* fc1_w = (const float*)d_in[7];
    const float* fc1_b = (const float*)d_in[8];
    const float* fc2_w = (const float*)d_in[9];
    const float* fc2_b = (const float*)d_in[10];
    const float* fs_w  = (const float*)d_in[11];
    const float* fs_b  = (const float*)d_in[12];
    float* out = (float*)d_out;

    // workspace carve-up (256B aligned)
    size_t off = 0;
    auto carve = [&](size_t bytes) {
        void* p = (char*)d_ws + off;
        off += (bytes + 255) & ~(size_t)255;
        return p;
    };
    _Float16* Xh   = (_Float16*)carve((size_t)MSEQ * KP * 2);   // 80 MB
    _Float16* W1t  = (_Float16*)carve((size_t)HH * KP * 2);
    _Float16* W2t  = (_Float16*)carve((size_t)HH * KP * 2);
    _Float16* px1  = (_Float16*)carve((size_t)MSEQ * HH * 2);   // 64 MB
    _Float16* px2  = (_Float16*)carve((size_t)MSEQ * HH * 2);   // 64 MB
    _Float16* hcat = (_Float16*)carve((size_t)BB * H2 * 2);
    _Float16* y1   = (_Float16*)carve((size_t)BB * H2 * 2);
    _Float16* y2   = (_Float16*)carve((size_t)BB * HH * 2);

    // 1) conversions
    {
        size_t tot = (size_t)MSEQ * KP;
        int blocks = (int)((tot + 255) / 256);
        cvt_x_kernel<<<blocks, 256, 0, stream>>>(seq, Xh);
        cvt_w_kernel<<<(2 * HH * KP + 255) / 256, 256, 0, stream>>>(W1x, W2x, W1t, W2t);
    }
    // 2) input projections (bias folded): one wave per 16x16 tile
    {
        int waves  = (MSEQ / 16) * (HH / 16);   // 131072
        int blocks = waves / 8;                 // 256 thr = 8 waves
        gemm_px_kernel<<<blocks, 256, 0, stream>>>(Xh, W1t, b1, px1);
        gemm_px_kernel<<<blocks, 256, 0, stream>>>(Xh, W2t, b2, px2);
    }
    // 3) both recurrence directions in one launch (batch-parallel chains)
    rnn_scan_kernel<<<dim3(BB / 16, 2), 256, 0, stream>>>(px1, px2, W1h, W2h, hcat);

    // 4) MLP head
    {
        int t1 = (BB / 16) * (H2 / 16);         // 512 tiles
        gemm_fc_kernel<<<t1 / 8, 256, 0, stream>>>(hcat, fc1_w, fc1_b, y1,
                                                   H2 / 16, H2, H2, t1, 1);
        int t2 = (BB / 16) * (HH / 16);         // 256 tiles
        gemm_fc_kernel<<<t2 / 8, 256, 0, stream>>>(y1, fc2_w, fc2_b, y2,
                                                   HH / 16, H2, HH, t2, 1);
        fs_head_kernel<<<1, 256, 0, stream>>>(y2, fs_w, fs_b, out);
    }
}
